// TemporalAttention_23450521436435
// MI455X (gfx1250) — compile-verified, matrix operand order checked by reading, not run on hardware
//
#include <hip/hip_runtime.h>
#include <stdint.h>

typedef __attribute__((ext_vector_type(16))) _Float16 v16h;
typedef __attribute__((ext_vector_type(8)))  float    v8f;
typedef __attribute__((ext_vector_type(2)))  _Float16 h2;
typedef __attribute__((ext_vector_type(2)))  float    f2;
typedef __attribute__((ext_vector_type(4)))  float    f4;

#define FEAT   128
#define TDIM   32
#define KVIN   160      // FEAT + TDIM, = 5 * 32 (exact K-loop)
#define NN     200
#define OUTD   64
#define NH     4
#define HD     16
#define CH     64       // neighbor chunk
#define NCHUNK 4
#define KV_STRIDE 164   // halves; 82 dwords -> conflict-friendly row stride
#define KC_STRIDE 66    // halves; 33 dwords -> odd stride, full bank period
#define VF_ROWS   208   // 200 rounded up to 16
#define KV_ROWS   (CH + VF_ROWS)   // unified buffer: K chunk rows [0,64), V rows [64,272)

__global__ void __launch_bounds__(256)
temporal_attn_kernel(const float* __restrict__ query,
                     const float* __restrict__ keys,
                     const float* __restrict__ times,
                     const uint8_t* __restrict__ mask,
                     const float* __restrict__ t2v_w0,
                     const float* __restrict__ t2v_b0,
                     const float* __restrict__ t2v_w,
                     const float* __restrict__ t2v_b,
                     const float* __restrict__ q_w, const float* __restrict__ q_b,
                     const float* __restrict__ k_w, const float* __restrict__ k_b,
                     const float* __restrict__ v_w, const float* __restrict__ v_b,
                     const float* __restrict__ o_w, const float* __restrict__ o_b,
                     float* __restrict__ out)
{
    __shared__ _Float16 kvin[CH * KV_STRIDE];        // 20992 B  (A-matrix chunk, f16)
    __shared__ _Float16 KVbuf[KV_ROWS * KC_STRIDE];  // 35904 B  (K chunk + full V)
    __shared__ float    Qs[OUTD];
    __shared__ float    logits[NH * VF_ROWS];        //  3328 B
    __shared__ float    ao[OUTD];

    const int b    = blockIdx.x;
    const int tid  = threadIdx.x;
    const int lane = tid & 31;
    const int wave = tid >> 5;

    // ---------------- per-wave weight (B-matrix) fragments, cached in VGPRs ----
    // waves 0-3: K projection ntile 0..3 ; waves 4-7: V projection ntile 0..3
    const bool  is_v  = (wave >= 4);
    const int   ntile = wave & 3;
    const float* wmat  = is_v ? v_w : k_w;           // [64][160] row-major (out x in)
    const float* wbias = is_v ? v_b : k_b;
    const int bcol  = ntile * 16 + (lane & 15);      // output column this lane owns
    const int khalf = (lane >= 16) ? 8 : 0;          // K sub-block per ISA layout
    const float cbias = wbias[bcol];

    v16h bfrag[5];
    #pragma unroll
    for (int kb = 0; kb < 5; ++kb) {
        const float* wr = wmat + (size_t)bcol * KVIN + kb * 32;
        #pragma unroll
        for (int p = 0; p < 8; ++p) {
            int kk = 2 * p + ((p >= 4) ? 8 : 0) + khalf;   // K(e,lane) mapping
            f2 w2 = *(const f2*)(wr + kk);
            bfrag[kb][2 * p]     = (_Float16)w2.x;
            bfrag[kb][2 * p + 1] = (_Float16)w2.y;
        }
    }

    // ---------------- Q projection: Qs[64] = q_w @ query[b] + q_b ----------------
    if (tid < OUTD) {
        const float* qrow = query + (size_t)b * FEAT;
        const float* wr   = q_w + (size_t)tid * FEAT;
        float acc = q_b[tid];
        #pragma unroll 4
        for (int k = 0; k < FEAT; k += 4) {
            f4 qv = *(const f4*)(qrow + k);
            f4 wv = *(const f4*)(wr + k);
            acc += qv.x * wv.x + qv.y * wv.y + qv.z * wv.z + qv.w * wv.w;
        }
        Qs[tid] = acc;
    }

    const float tw0 = t2v_w0[0], tb0 = t2v_b0[0];

    for (int ch = 0; ch < NCHUNK; ++ch) {
        const int n0 = ch * CH;
        const int vrow_base = is_v ? (CH + n0) : 0;  // unified KVbuf row offset
        __syncthreads();   // prior chunk's kvin/K consumers done

        // ---- stage kv_input chunk into LDS as f16: keys || time2vec ----
        for (int i = 0; i < 8; ++i) {              // 64 rows x 32 float4
            int idx = tid + i * 256;
            int row = idx >> 5;
            int c4  = idx & 31;
            int ng  = n0 + row;
            f4 kv = {0.f, 0.f, 0.f, 0.f};
            if (ng < NN)
                kv = *(const f4*)(keys + ((size_t)b * NN + ng) * FEAT + c4 * 4);
            h2 lo = { (_Float16)kv.x, (_Float16)kv.y };
            h2 hi = { (_Float16)kv.z, (_Float16)kv.w };
            _Float16* dst = kvin + row * KV_STRIDE + c4 * 4;
            *(h2*)(dst)     = lo;
            *(h2*)(dst + 2) = hi;
        }
        for (int i = 0; i < 8; ++i) {              // 64 rows x 32 time channels
            int idx = tid + i * 256;
            int row = idx >> 5;
            int j   = idx & 31;
            int ng  = n0 + row;
            float val = 0.f;
            if (ng < NN) {
                float t = times[(size_t)b * NN + ng];
                val = (j == 0) ? (t * tw0 + tb0)
                               : __sinf(t * t2v_w[j - 1] + t2v_b[j - 1]);
            }
            kvin[row * KV_STRIDE + FEAT + j] = (_Float16)val;
        }
        // prefetch next chunk's keys rows while we compute on this one
        if (ch + 1 < NCHUNK) {
            int nrow = n0 + CH + (tid >> 2);                 // 4 threads / row
            if (nrow < NN) {
                const float* p = keys + ((size_t)b * NN + nrow) * FEAT + (tid & 3) * 32;
                __builtin_prefetch(p, 0, 3);
            }
        }
        __syncthreads();

        // ---- WMMA projections: D = kvin[64x160] x W^T[160x64] (+bias) ----
        // Reverse mt order + clamped row base => branch-free stores. The only
        // clamped tiles are V tiles in the last chunk (rows 208..255, never
        // read); they land on rows 256..271 and are overwritten afterwards by
        // the valid mt=0 tile (V rows 192..207 == rows 256..271 exactly).
        const int arow = lane & 15;
        for (int mt = 3; mt >= 0; --mt) {
            const _Float16* ar = kvin + (mt * 16 + arow) * KV_STRIDE;
            // load ALL five A fragments first (one dscnt chain), then 5 WMMAs b2b
            v16h a[5];
            #pragma unroll
            for (int kb = 0; kb < 5; ++kb) {
                #pragma unroll
                for (int p = 0; p < 8; ++p) {
                    int kk = kb * 32 + 2 * p + ((p >= 4) ? 8 : 0) + khalf;
                    h2 d = *(const h2*)(ar + kk);
                    a[kb][2 * p]     = d.x;
                    a[kb][2 * p + 1] = d.y;
                }
            }
            v8f c = {};
            #pragma unroll
            for (int kb = 0; kb < 5; ++kb)
                c = __builtin_amdgcn_wmma_f32_16x16x32_f16(
                        false, a[kb], false, bfrag[kb], (short)0, c, false, false);

            int rowbase = vrow_base + mt * 16;
            rowbase = rowbase > (KV_ROWS - 16) ? (KV_ROWS - 16) : rowbase;  // v_min
            _Float16* dbase = KVbuf + (rowbase + ((lane >= 16) ? 8 : 0)) * KC_STRIDE + bcol;
            #pragma unroll
            for (int r = 0; r < 8; ++r)            // C/D: lanes16-31 -> M=r+8
                dbase[r * KC_STRIDE] = (_Float16)(c[r] + cbias);
        }
        __syncthreads();

        // ---- masked logits for this chunk: tid -> (h = tid/64, n = tid%64) ----
        {
            int h  = tid >> 6;
            int n  = tid & 63;
            int ng = n0 + n;
            if (ng < VF_ROWS) {
                float logit = -__builtin_inff();
                if (ng < NN && mask[(size_t)b * NN + ng]) {
                    const _Float16* kr = KVbuf + n * KC_STRIDE + h * HD;  // K rows [0,64)
                    const float*    qh = Qs + h * HD;
                    float acc = 0.f;
                    #pragma unroll
                    for (int d = 0; d < HD; ++d)
                        acc += qh[d] * (float)kr[d];
                    logit = acc * 0.25f;           // / sqrt(HEAD_DIM=16)
                }
                logits[h * VF_ROWS + ng] = logit;
            }
        }
    }
    __syncthreads();

    // ---------------- softmax: wave w < 4 handles head w (wave32 shuffles) ------
    if (wave < NH) {
        float* lrow = logits + wave * VF_ROWS;
        float mx = -__builtin_inff();
        for (int n = lane; n < VF_ROWS; n += 32)
            mx = fmaxf(mx, lrow[n]);
        #pragma unroll
        for (int off = 16; off > 0; off >>= 1)
            mx = fmaxf(mx, __shfl_xor(mx, off, 32));
        float sum = 0.f;
        for (int n = lane; n < VF_ROWS; n += 32)
            sum += __expf(lrow[n] - mx);           // exp(-inf)=0 handles mask/pad
        #pragma unroll
        for (int off = 16; off > 0; off >>= 1)
            sum += __shfl_xor(sum, off, 32);
        float inv = 1.f / sum;
        for (int n = lane; n < VF_ROWS; n += 32)
            lrow[n] = __expf(lrow[n] - mx) * inv;  // recompute: avoids scratch array
    }
    __syncthreads();

    // ---------------- attn @ V : thread t<64 -> (h = t/16, d = t%16) ------------
    if (tid < OUTD) {
        int h = tid >> 4, d = tid & 15;
        const float* prow = logits + h * VF_ROWS;
        const _Float16* vcol = KVbuf + CH * KC_STRIDE + h * HD + d;  // V rows [64,272)
        float acc = 0.f;
        for (int n = 0; n < NN; ++n)
            acc += prow[n] * (float)vcol[n * KC_STRIDE];
        ao[tid] = acc;
    }
    __syncthreads();

    // ---------------- output projection ----------------------------------------
    if (tid < OUTD) {
        const float* wr = o_w + (size_t)tid * OUTD;
        float acc = o_b[tid];
        #pragma unroll 4
        for (int k = 0; k < OUTD; ++k)
            acc += ao[k] * wr[k];
        out[(size_t)b * OUTD + tid] = acc;
    }
}

extern "C" void kernel_launch(void* const* d_in, const int* in_sizes, int n_in,
                              void* d_out, int out_size, void* d_ws, size_t ws_size,
                              hipStream_t stream) {
    const float*   query  = (const float*)d_in[0];
    const float*   keys   = (const float*)d_in[1];
    const float*   times  = (const float*)d_in[2];
    const uint8_t* mask   = (const uint8_t*)d_in[3];   // jnp bool -> 1 byte/elem
    const float*   t2v_w0 = (const float*)d_in[4];
    const float*   t2v_b0 = (const float*)d_in[5];
    const float*   t2v_w  = (const float*)d_in[6];
    const float*   t2v_b  = (const float*)d_in[7];
    const float*   q_w    = (const float*)d_in[8];
    const float*   q_b    = (const float*)d_in[9];
    const float*   k_w    = (const float*)d_in[10];
    const float*   k_b    = (const float*)d_in[11];
    const float*   v_w    = (const float*)d_in[12];
    const float*   v_b    = (const float*)d_in[13];
    const float*   o_w    = (const float*)d_in[14];
    const float*   o_b    = (const float*)d_in[15];

    const int B = in_sizes[0] / FEAT;                  // 4096

    temporal_attn_kernel<<<B, 256, 0, stream>>>(
        query, keys, times, mask,
        t2v_w0, t2v_b0, t2v_w, t2v_b,
        q_w, q_b, k_w, k_b, v_w, v_b, o_w, o_b,
        (float*)d_out);
}
